// TernaryDense_66013647339837
// MI455X (gfx1250) — compile-verified, matrix operand order checked by reading, not run on hardware
//
#include <hip/hip_runtime.h>

typedef __attribute__((ext_vector_type(16))) __bf16 v16bf;
typedef __attribute__((ext_vector_type(8)))  __bf16 v8bf;
typedef __attribute__((ext_vector_type(4)))  __bf16 v4bf;
typedef __attribute__((ext_vector_type(8)))  float  v8f;

#define BATCH   8192
#define N_IN    4096
#define N_UNITS 4096
#define W_ELEMS (N_IN * N_UNITS)

#define BM 128
#define BN 128
#define BK 32
#define LDA 40          // padded LDS stride (bf16 elems): 80 B, 16B-aligned
#define NPART 1024

// ---------------- deterministic reductions over W ----------------

__global__ __launch_bounds__(256) void k_abs_partial(const float* __restrict__ W,
                                                     float* __restrict__ part) {
    __shared__ float sm[256];
    float s = 0.f;
    for (int i = blockIdx.x * 256 + threadIdx.x; i < W_ELEMS; i += 256 * NPART)
        s += fabsf(W[i]);
    sm[threadIdx.x] = s;
    __syncthreads();
    for (int off = 128; off > 0; off >>= 1) {
        if (threadIdx.x < (unsigned)off) sm[threadIdx.x] += sm[threadIdx.x + off];
        __syncthreads();
    }
    if (threadIdx.x == 0) part[blockIdx.x] = sm[0];
}

__global__ __launch_bounds__(256) void k_abs_final(const float* __restrict__ part,
                                                   float* __restrict__ ws) {
    __shared__ float sm[256];
    float s = 0.f;
    for (int i = threadIdx.x; i < NPART; i += 256) s += part[i];
    sm[threadIdx.x] = s;
    __syncthreads();
    for (int off = 128; off > 0; off >>= 1) {
        if (threadIdx.x < (unsigned)off) sm[threadIdx.x] += sm[threadIdx.x + off];
        __syncthreads();
    }
    if (threadIdx.x == 0) ws[0] = sm[0];   // sum |W|
}

__global__ __launch_bounds__(256) void k_alpha_partial(const float* __restrict__ W,
                                                       const float* __restrict__ ws,
                                                       float* __restrict__ partM,
                                                       float* __restrict__ partC) {
    const float t = 0.7f * ws[0] * (1.0f / (float)W_ELEMS);
    __shared__ float smM[256];
    __shared__ float smC[256];
    float sM = 0.f, sC = 0.f;
    for (int i = blockIdx.x * 256 + threadIdx.x; i < W_ELEMS; i += 256 * NPART) {
        float w = W[i];
        float a = fabsf(w);
        if (a > t) { sM += a; sC += 1.0f; }   // strict, matches reference
    }
    smM[threadIdx.x] = sM;
    smC[threadIdx.x] = sC;
    __syncthreads();
    for (int off = 128; off > 0; off >>= 1) {
        if (threadIdx.x < (unsigned)off) {
            smM[threadIdx.x] += smM[threadIdx.x + off];
            smC[threadIdx.x] += smC[threadIdx.x + off];
        }
        __syncthreads();
    }
    if (threadIdx.x == 0) { partM[blockIdx.x] = smM[0]; partC[blockIdx.x] = smC[0]; }
}

__global__ __launch_bounds__(256) void k_alpha_final(const float* __restrict__ partM,
                                                     const float* __restrict__ partC,
                                                     float* __restrict__ ws) {
    __shared__ float smM[256];
    __shared__ float smC[256];
    float sM = 0.f, sC = 0.f;
    for (int i = threadIdx.x; i < NPART; i += 256) { sM += partM[i]; sC += partC[i]; }
    smM[threadIdx.x] = sM;
    smC[threadIdx.x] = sC;
    __syncthreads();
    for (int off = 128; off > 0; off >>= 1) {
        if (threadIdx.x < (unsigned)off) {
            smM[threadIdx.x] += smM[threadIdx.x + off];
            smC[threadIdx.x] += smC[threadIdx.x + off];
        }
        __syncthreads();
    }
    if (threadIdx.x == 0) { ws[1] = smM[0]; ws[2] = smC[0]; }
}

// ---------------- bf16-split WMMA GEMM ----------------
// out[8192,4096] = alpha * (A @ Wt) + bias ; A split into bf16 hi+lo, Wt exact bf16 ternary.

__global__ __launch_bounds__(256) void k_ternary_gemm(const float* __restrict__ A,
                                                      const float* __restrict__ W,
                                                      const float* __restrict__ bias,
                                                      const float* __restrict__ ws,
                                                      float* __restrict__ out) {
    __shared__ __align__(16) __bf16 Ah[2][BM][LDA];
    __shared__ __align__(16) __bf16 Al[2][BM][LDA];
    __shared__ __align__(16) __bf16 Bs[2][BN][LDA];   // column-major: Bs[buf][n][k]

    const float t     = 0.7f * ws[0] * (1.0f / (float)W_ELEMS);
    const float alpha = ws[1] / ws[2];

    const int tid  = threadIdx.x;
    const int lane = tid & 31;
    const int wave = tid >> 5;
    const int wm   = wave >> 1;     // 0..3 : 32-row slab
    const int wn   = wave & 1;      // 0..1 : 64-col slab
    const int MB   = blockIdx.y * BM;
    const int NB   = blockIdx.x * BN;

    // staging registers for the next K-tile
    float4 aReg[4];
    float  wReg[16];
    const int wcol = tid & 127;     // column within B tile
    const int wkh  = tid >> 7;      // 0/1 : which 16-k half

    auto load_global = [&](int kb) {
#pragma unroll
        for (int i = 0; i < 4; ++i) {
            int f = tid + i * 256;                 // 1024 float4 per A tile
            int row = f >> 3, kq = (f & 7) * 4;
            aReg[i] = *(const float4*)(A + (size_t)(MB + row) * N_IN + kb + kq);
        }
#pragma unroll
        for (int j = 0; j < 16; ++j) {
            int k = kb + wkh * 16 + j;
            wReg[j] = W[(size_t)k * N_UNITS + NB + wcol];
        }
        if (kb + 2 * BK < N_IN) {  // global_prefetch_b8 for tile k+2
            __builtin_prefetch(A + (size_t)(MB + (tid >> 3)) * N_IN + kb + 2 * BK, 0, 1);
            __builtin_prefetch(W + (size_t)(kb + 2 * BK + wkh * 16) * N_UNITS + NB + wcol, 0, 1);
        }
    };

    auto store_lds = [&](int buf) {
#pragma unroll
        for (int i = 0; i < 4; ++i) {
            int f = tid + i * 256;
            int row = f >> 3, kq = (f & 7) * 4;
            float x[4] = {aReg[i].x, aReg[i].y, aReg[i].z, aReg[i].w};
            v4bf hi, lo;
#pragma unroll
            for (int e = 0; e < 4; ++e) {
                __bf16 h = (__bf16)x[e];
                hi[e] = h;
                lo[e] = (__bf16)(x[e] - (float)h);
            }
            *(v4bf*)&Ah[buf][row][kq] = hi;
            *(v4bf*)&Al[buf][row][kq] = lo;
        }
        v8bf t0, t1;
#pragma unroll
        for (int j = 0; j < 16; ++j) {
            float w = wReg[j];
            __bf16 v = (w > t) ? (__bf16)1.0f : ((w < -t) ? (__bf16)-1.0f : (__bf16)0.0f);
            if (j < 8) t0[j] = v; else t1[j - 8] = v;
        }
        *(v8bf*)&Bs[buf][wcol][wkh * 16]     = t0;
        *(v8bf*)&Bs[buf][wcol][wkh * 16 + 8] = t1;
    };

    // 16-bit A/B fragment load per ISA layout:
    // lanes 0-15 hold K{0..7,16..23}, lanes 16-31 hold K{8..15,24..31} of row (lane&15)+base
    auto load_frag = [&](const __bf16 (*S)[LDA], int rbase) -> v16bf {
        int r  = rbase + (lane & 15);
        int kh = (lane >> 4) * 8;
        v8bf p0 = *(const v8bf*)&S[r][kh];
        v8bf p1 = *(const v8bf*)&S[r][16 + kh];
        return __builtin_shufflevector(p0, p1, 0, 1, 2, 3, 4, 5, 6, 7,
                                               8, 9, 10, 11, 12, 13, 14, 15);
    };

    v8f acc[2][4];
#pragma unroll
    for (int i = 0; i < 2; ++i)
#pragma unroll
        for (int j = 0; j < 4; ++j) acc[i][j] = (v8f){};

    // prologue
    load_global(0);
    store_lds(0);
    __syncthreads();

    const int KT = N_IN / BK;   // 128
    for (int kt = 0; kt < KT; ++kt) {
        int cur = kt & 1;
        if (kt + 1 < KT) load_global((kt + 1) * BK);   // hide latency behind WMMAs

        v16bf ah[2], al[2], bb[4];
#pragma unroll
        for (int i = 0; i < 2; ++i) {
            ah[i] = load_frag(Ah[cur], wm * 32 + i * 16);
            al[i] = load_frag(Al[cur], wm * 32 + i * 16);
        }
#pragma unroll
        for (int j = 0; j < 4; ++j) bb[j] = load_frag(Bs[cur], wn * 64 + j * 16);

#pragma unroll
        for (int i = 0; i < 2; ++i)
#pragma unroll
            for (int j = 0; j < 4; ++j) {
                acc[i][j] = __builtin_amdgcn_wmma_f32_16x16x32_bf16(
                    false, ah[i], false, bb[j], (short)0, acc[i][j], false, false);
                acc[i][j] = __builtin_amdgcn_wmma_f32_16x16x32_bf16(
                    false, al[i], false, bb[j], (short)0, acc[i][j], false, false);
            }

        if (kt + 1 < KT) store_lds(cur ^ 1);
        __syncthreads();
    }

    // epilogue: C layout — VGPR r, lane l: row = base + r + (l>=16)*8, col = base + (l&15)
    const int col0 = NB + wn * 64;
#pragma unroll
    for (int i = 0; i < 2; ++i) {
        int rowb = MB + wm * 32 + i * 16 + (lane >> 4) * 8;
        int c = lane & 15;
#pragma unroll
        for (int j = 0; j < 4; ++j) {
            int col = col0 + j * 16 + c;
            float bv = bias[col];
#pragma unroll
            for (int r = 0; r < 8; ++r) {
                int row = rowb + r;
                out[(size_t)row * N_UNITS + col] = alpha * acc[i][j][r] + bv;
            }
        }
    }
}

// ---------------- host launch ----------------

extern "C" void kernel_launch(void* const* d_in, const int* in_sizes, int n_in,
                              void* d_out, int out_size, void* d_ws, size_t ws_size,
                              hipStream_t stream) {
    const float* A    = (const float*)d_in[0];   // [8192, 4096]
    const float* W    = (const float*)d_in[1];   // [4096, 4096]
    const float* bias = (const float*)d_in[2];   // [4096]
    float*       out  = (float*)d_out;           // [8192, 4096]

    float* wsf   = (float*)d_ws;
    float* partA = wsf + 16;                     // [NPART]
    float* partM = partA + NPART;                // [NPART]
    float* partC = partM + NPART;                // [NPART]

    k_abs_partial <<<NPART, 256, 0, stream>>>(W, partA);
    k_abs_final   <<<1,     256, 0, stream>>>(partA, wsf);
    k_alpha_partial<<<NPART, 256, 0, stream>>>(W, wsf, partM, partC);
    k_alpha_final <<<1,     256, 0, stream>>>(partM, partC, wsf);

    dim3 grid(N_UNITS / BN, BATCH / BM);         // (32, 64)
    k_ternary_gemm<<<grid, 256, 0, stream>>>(A, W, bias, wsf, out);
}
